// SetOfSetLayerSelfAttention_70531952934916
// MI455X (gfx1250) — matrix operand order repaired and checked by reading, not compiled, not run on hardware
//
#include <hip/hip_runtime.h>
#include <hip/hip_bf16.h>

// Problem constants (from the reference)
#define NNZ_   262144
#define M_     256
#define N_     8192
#define D_     256

// Padded LDS strides (elements) to avoid 64-bank conflicts on fragment reads
#define WL  264   // for 256-wide tiles (264*2B = 528B row, 132 dwords -> bank +4/row)
#define VL  40    // for 32-wide tiles  (40*2B = 80B row, 20 dwords -> bank +20/row)

typedef __attribute__((ext_vector_type(16))) __bf16 v16bf;
typedef __attribute__((ext_vector_type(8)))  __bf16 v8bf;
typedef __attribute__((ext_vector_type(4)))  __bf16 v4bf;
typedef __attribute__((ext_vector_type(8)))  float  v8f;

// ---------------------------------------------------------------------------
// WMMA helpers (CDNA5 v_wmma_f32_16x16x32_bf16, wave32)
// A layout (16-bit A 16x32, ISA 7.12.2): lane -> m = l&15, half = l>>4;
//   per lane the fragment is TWO contiguous 8-elem chunks:
//     k = k_base + 8*half + [0..7]  and  k = k_base + 16 + 8*half + [0..7]
// B layout (32x16): lane -> n = l&15; same two-chunk k order as A
//   (8-elem chunks at +8*half and +16+8*half of a k-contiguous-per-column source).
// C/D: lane -> n = l&15, vgpr r -> m = r + 8*(l>>4).
// ---------------------------------------------------------------------------
__device__ __forceinline__ v8f wmma_bf16(v16bf a, v16bf b, v8f c) {
    return __builtin_amdgcn_wmma_f32_16x16x32_bf16(false, a, false, b, (short)0, c,
                                                   false, false);
}

__device__ __forceinline__ v16bf cat8(v8bf lo, v8bf hi) {
    return __builtin_shufflevector(lo, hi, 0, 1, 2, 3, 4, 5, 6, 7,
                                   8, 9, 10, 11, 12, 13, 14, 15);
}

// A-fragment: rows are m (row_base+m), k-contiguous storage with stride ld
__device__ __forceinline__ v16bf lds_a_frag(const __bf16* s, int ld, int row_base,
                                            int k_base, int lane) {
    int half = lane >> 4, m = lane & 15;
    const __bf16* p = s + (row_base + m) * ld + k_base + 8 * half;
    return cat8(*(const v8bf*)p, *(const v8bf*)(p + 16));  // 2x ds_load_b128
}

// B-fragment from a k-contiguous-per-column source: row index = n, stride ld
__device__ __forceinline__ v16bf lds_b_frag(const __bf16* s, int ld, int n_base,
                                            int k_base, int lane) {
    int half = lane >> 4, n = lane & 15;
    const __bf16* p = s + (n_base + n) * ld + k_base + 8 * half;
    return cat8(*(const v8bf*)p, *(const v8bf*)(p + 16));  // 2x ds_load_b128
}

__device__ __forceinline__ void st4bf(__bf16* dst, float4 v) {
    v4bf t = {(__bf16)v.x, (__bf16)v.y, (__bf16)v.z, (__bf16)v.w};
    *(v4bf*)dst = t;                                        // ds_store_b64
}

// ---------------------------------------------------------------------------
// 1) zero accumulators
// ---------------------------------------------------------------------------
__global__ void zero_f32(float* __restrict__ p, int n) {
    int i = blockIdx.x * blockDim.x + threadIdx.x;
    if (i < n) p[i] = 0.f;
}

// ---------------------------------------------------------------------------
// 2) segment sums via f32 global atomics (bandwidth-dominated: 256 MB read)
// ---------------------------------------------------------------------------
__global__ __launch_bounds__(256) void seg_accum(
    const float* __restrict__ values, const int* __restrict__ row_idx,
    const int* __restrict__ col_idx, float* __restrict__ sum_rows,
    float* __restrict__ cnt_rows, float* __restrict__ sum_cols,
    float* __restrict__ cnt_cols) {
    const int obs = blockIdx.x;
    const int d = threadIdx.x;
    float v = values[(size_t)obs * D_ + d];
    int c = col_idx[obs], r = row_idx[obs];
    atomicAdd(sum_rows + (size_t)c * D_ + d, v);
    atomicAdd(sum_cols + (size_t)r * D_ + d, v);
    if (d == 0) {
        atomicAdd(cnt_rows + c, 1.0f);
        atomicAdd(cnt_cols + r, 1.0f);
    }
}

// ---------------------------------------------------------------------------
// 3) global mean (from RAW row-index sums) + out_both = g@W_both + b_both
// ---------------------------------------------------------------------------
__global__ __launch_bounds__(256) void global_and_both(
    const float* __restrict__ sum_cols_raw, const float* __restrict__ W_both,
    const float* __restrict__ b_both, float* __restrict__ out_both) {
    __shared__ float sg[D_];
    int t = threadIdx.x;
    float s = 0.f;
    for (int m = 0; m < M_; ++m) s += sum_cols_raw[m * D_ + t];
    sg[t] = s * (1.0f / (float)NNZ_);
    __syncthreads();
    float o = b_both[t];
    for (int d = 0; d < D_; ++d) o += sg[d] * W_both[d * D_ + t];
    out_both[t] = o;
}

// ---------------------------------------------------------------------------
// 4) sums -> means (divide by max(count,1)), in place
// ---------------------------------------------------------------------------
__global__ void seg_normalize(float* __restrict__ sum, const float* __restrict__ cnt,
                              int rows) {
    int i = blockIdx.x * blockDim.x + threadIdx.x;
    if (i < rows * D_) {
        int r = i >> 8;
        sum[i] /= fmaxf(cnt[r], 1.0f);
    }
}

// ---------------------------------------------------------------------------
// Shared staging helpers (256-thread blocks)
// ---------------------------------------------------------------------------
// Stage a [rows x 256] fp32 tile row-major into LDS bf16 (stride WL).
__device__ __forceinline__ void stage_rowmajor(const float* __restrict__ g,
                                               size_t row0, int rows,
                                               __bf16* __restrict__ s, int tid) {
    for (int i = tid; i < rows * 64; i += 256) {
        int r = i >> 6, c4 = (i & 63) * 4;
        st4bf(s + r * WL + c4, *(const float4*)(g + (row0 + r) * D_ + c4));
    }
}

// Stage W[256x256] TRANSPOSED into LDS: sWt[n*WL + k] = W[k*256 + n].
// Global reads stay coalesced (consecutive threads -> consecutive n).
__device__ __forceinline__ void stage_w_transposed(const float* __restrict__ W,
                                                   __bf16* __restrict__ sWt,
                                                   int tid) {
    for (int i = tid; i < 64 * 256; i += 256) {
        int n = i & 255, k = (i >> 8) * 4;
        v4bf t = {(__bf16)W[(k + 0) * D_ + n], (__bf16)W[(k + 1) * D_ + n],
                  (__bf16)W[(k + 2) * D_ + n], (__bf16)W[(k + 3) * D_ + n]};
        *(v4bf*)(sWt + n * WL + k) = t;  // 8B-aligned ds_store_b64
    }
}

// ---------------------------------------------------------------------------
// 5) C[R,256] = A[R,256] @ W[256,256] + bias  (bf16 WMMA, f32 accumulate)
//    block = 256 threads = 8 waves; each wave owns a 32x256 output strip
//    (two 16-row A strips): every B-fragment feeds TWO independent WMMA
//    accumulator chains. All 8 B-fragments of an n-tile are preloaded into
//    distinct VGPRs (one ds_load_b128 clause) so the WMMAs retire against
//    partial dscnt waits instead of stalling on each pair of loads.
// ---------------------------------------------------------------------------
__global__ __launch_bounds__(256) void gemm_bias_wmma(
    const float* __restrict__ A, const float* __restrict__ W,
    const float* __restrict__ bias, float* __restrict__ C) {
    extern __shared__ char smem[];
    __bf16* sWt = (__bf16*)smem;          // 256 x WL (transposed W)
    __bf16* sA = sWt + D_ * WL;           // 256 x WL
    const int tid = threadIdx.x;
    const size_t row0 = (size_t)blockIdx.x * 256;
    stage_w_transposed(W, sWt, tid);
    stage_rowmajor(A, row0, 256, sA, tid);
    __syncthreads();
    const int lane = tid & 31, wave = tid >> 5;
    const int half = lane >> 4, n16 = lane & 15;
    v16bf af0[8], af1[8];
#pragma unroll
    for (int kk = 0; kk < 8; ++kk) {
        af0[kk] = lds_a_frag(sA, WL, wave * 32, kk * 32, lane);
        af1[kk] = lds_a_frag(sA, WL, wave * 32 + 16, kk * 32, lane);
    }
#pragma unroll 1
    for (int nt = 0; nt < 16; ++nt) {
        v16bf bfr[8];
#pragma unroll
        for (int kk = 0; kk < 8; ++kk)
            bfr[kk] = lds_b_frag(sWt, WL, nt * 16, kk * 32, lane);
        v8f acc0 = {0.f, 0.f, 0.f, 0.f, 0.f, 0.f, 0.f, 0.f};
        v8f acc1 = {0.f, 0.f, 0.f, 0.f, 0.f, 0.f, 0.f, 0.f};
#pragma unroll
        for (int kk = 0; kk < 8; ++kk) {
            acc0 = wmma_bf16(af0[kk], bfr[kk], acc0);
            acc1 = wmma_bf16(af1[kk], bfr[kk], acc1);
        }
        float bv = bias[nt * 16 + n16];
#pragma unroll
        for (int r = 0; r < 8; ++r) {
            size_t rr = row0 + wave * 32 + r + 8 * half;
            C[rr * D_ + nt * 16 + n16] = acc0[r] + bv;
            C[(rr + 16) * D_ + nt * 16 + n16] = acc1[r] + bv;
        }
    }
}

// ---------------------------------------------------------------------------
// 6) Flash attention: O = softmax(Q K^T / 16) V, streaming over keys (the
//    nk x nk score matrix never hits HBM). Q tile staged once; K (row-major)
//    and V^T staged per 32-key block; P routed via per-wave LDS scratch with
//    s_wait_dscnt for wave-local DS ordering; next block prefetched.
// ---------------------------------------------------------------------------
__global__ __launch_bounds__(256) void flash_attn_wmma(
    const float* __restrict__ Q, const float* __restrict__ K,
    const float* __restrict__ V, float* __restrict__ O, int nk) {
    extern __shared__ char smem[];
    __bf16* sK = (__bf16*)smem;          // 32 x WL   (keys x d, row-major)
    __bf16* sVt = sK + 32 * WL;          // 256 x VL  (d x keys, transposed)
    __bf16* sQ = sVt + D_ * VL;          // 128 x WL
    __bf16* sP = sQ + 128 * WL;          // 8 waves x (16 x VL)
    const int tid = threadIdx.x, lane = tid & 31, wave = tid >> 5;
    const int half = lane >> 4, l16 = lane & 15;
    const size_t qblk = (size_t)blockIdx.x * 128;
    const int qrow0 = (int)qblk + wave * 16;
    __bf16* myP = sP + wave * 16 * VL;

    stage_rowmajor(Q, qblk, 128, sQ, tid);
    __syncthreads();
    v16bf qf[8];
#pragma unroll
    for (int kk = 0; kk < 8; ++kk)
        qf[kk] = lds_a_frag(sQ, WL, wave * 16, kk * 32, lane);

    v8f oacc[16];
#pragma unroll
    for (int ct = 0; ct < 16; ++ct)
#pragma unroll
        for (int r = 0; r < 8; ++r) oacc[ct][r] = 0.f;
    float mmax[8], lsum[8];
#pragma unroll
    for (int r = 0; r < 8; ++r) { mmax[r] = -3.0e38f; lsum[r] = 0.f; }

    for (int kb = 0; kb < nk; kb += 32) {
        // stage K block row-major
        for (int i = tid; i < 32 * 64; i += 256) {
            int r = i >> 6, c4 = (i & 63) * 4;
            st4bf(sK + r * WL + c4, *(const float4*)(K + (size_t)(kb + r) * D_ + c4));
        }
        // stage V block transposed: sVt[d*VL + key] = V[(kb+key)*256 + d]
        for (int i = tid; i < 8 * 256; i += 256) {
            int d = i & 255, key = (i >> 8) * 4;
            v4bf t = {(__bf16)V[(size_t)(kb + key + 0) * D_ + d],
                      (__bf16)V[(size_t)(kb + key + 1) * D_ + d],
                      (__bf16)V[(size_t)(kb + key + 2) * D_ + d],
                      (__bf16)V[(size_t)(kb + key + 3) * D_ + d]};
            *(v4bf*)(sVt + d * VL + key) = t;
        }
        if (kb + 32 < nk) {  // hint next block into cache (global_prefetch_b8)
            __builtin_prefetch(K + (size_t)(kb + 32) * D_ + tid * 32, 0, 0);
            __builtin_prefetch(V + (size_t)(kb + 32) * D_ + tid * 32, 0, 0);
        }
        __syncthreads();

        // S(16x32) = Q Kblock^T : two independent 16x16 WMMA chains
        v8f s0 = {0.f, 0.f, 0.f, 0.f, 0.f, 0.f, 0.f, 0.f};
        v8f s1 = {0.f, 0.f, 0.f, 0.f, 0.f, 0.f, 0.f, 0.f};
#pragma unroll
        for (int kk = 0; kk < 8; ++kk) {
            s0 = wmma_bf16(qf[kk], lds_b_frag(sK, WL, 0, kk * 32, lane), s0);
            s1 = wmma_bf16(qf[kk], lds_b_frag(sK, WL, 16, kk * 32, lane), s1);
        }

        // online softmax: row stats live on the 16 lanes of each half
        float alpha8[8];
#pragma unroll
        for (int r = 0; r < 8; ++r) {
            float x0 = s0[r] * 0.0625f, x1 = s1[r] * 0.0625f;  // 1/sqrt(256)
            float bm = fmaxf(x0, x1);
#pragma unroll
            for (int off = 1; off < 16; off <<= 1)
                bm = fmaxf(bm, __shfl_xor(bm, off, 32));
            float mnew = fmaxf(mmax[r], bm);
            float p0 = __expf(x0 - mnew), p1 = __expf(x1 - mnew);
            float rs = p0 + p1;
#pragma unroll
            for (int off = 1; off < 16; off <<= 1) rs += __shfl_xor(rs, off, 32);
            alpha8[r] = __expf(mmax[r] - mnew);
            lsum[r] = lsum[r] * alpha8[r] + rs;
            mmax[r] = mnew;
            myP[(r + 8 * half) * VL + l16] = (__bf16)p0;
            myP[(r + 8 * half) * VL + 16 + l16] = (__bf16)p1;
        }
        // wave-local DS ordering before re-reading P as an A-fragment
        asm volatile("s_wait_dscnt 0" ::: "memory");
        v16bf pf = lds_a_frag(myP, VL, 0, 0, lane);

        // O(16x256) = alpha*O + P(16x32) @ V(32x256): 16 independent chains
#pragma unroll
        for (int ct = 0; ct < 16; ++ct) {
#pragma unroll
            for (int r = 0; r < 8; ++r) oacc[ct][r] *= alpha8[r];
            oacc[ct] = wmma_bf16(pf, lds_b_frag(sVt, VL, ct * 16, 0, lane), oacc[ct]);
        }
        __syncthreads();
    }

#pragma unroll
    for (int r = 0; r < 8; ++r) lsum[r] = fmaxf(lsum[r], 1e-30f);
#pragma unroll
    for (int ct = 0; ct < 16; ++ct)
#pragma unroll
        for (int r = 0; r < 8; ++r)
            O[(size_t)(qrow0 + r + 8 * half) * D_ + ct * 16 + l16] =
                oacc[ct][r] / lsum[r];
}

// ---------------------------------------------------------------------------
// 7) Fused: out = (values@W_all + b_all + out_rows[col] + out_cols[row]
//                  + out_both) / 4   -- values read once, out written once.
//    Same 32-rows-per-wave, B-preloaded WMMA structure as the GEMM.
// ---------------------------------------------------------------------------
__global__ __launch_bounds__(256) void fused_combine_wmma(
    const float* __restrict__ values, const int* __restrict__ row_idx,
    const int* __restrict__ col_idx, const float* __restrict__ W_all,
    const float* __restrict__ b_all, const float* __restrict__ out_rows,
    const float* __restrict__ out_cols, const float* __restrict__ out_both,
    float* __restrict__ out) {
    extern __shared__ char smem[];
    __bf16* sWt = (__bf16*)smem;
    __bf16* sA = sWt + D_ * WL;
    const int tid = threadIdx.x;
    const size_t row0 = (size_t)blockIdx.x * 256;
    stage_w_transposed(W_all, sWt, tid);
    stage_rowmajor(values, row0, 256, sA, tid);
    __syncthreads();
    const int lane = tid & 31, wave = tid >> 5;
    const int half = lane >> 4, n16 = lane & 15;
    v16bf af0[8], af1[8];
#pragma unroll
    for (int kk = 0; kk < 8; ++kk) {
        af0[kk] = lds_a_frag(sA, WL, wave * 32, kk * 32, lane);
        af1[kk] = lds_a_frag(sA, WL, wave * 32 + 16, kk * 32, lane);
    }
    int cidx0[8], ridx0[8], cidx1[8], ridx1[8];
#pragma unroll
    for (int r = 0; r < 8; ++r) {
        size_t obs = row0 + wave * 32 + r + 8 * half;
        cidx0[r] = col_idx[obs];
        ridx0[r] = row_idx[obs];
        cidx1[r] = col_idx[obs + 16];
        ridx1[r] = row_idx[obs + 16];
    }
#pragma unroll 1
    for (int nt = 0; nt < 16; ++nt) {
        v16bf bfr[8];
#pragma unroll
        for (int kk = 0; kk < 8; ++kk)
            bfr[kk] = lds_b_frag(sWt, WL, nt * 16, kk * 32, lane);
        v8f acc0 = {0.f, 0.f, 0.f, 0.f, 0.f, 0.f, 0.f, 0.f};
        v8f acc1 = {0.f, 0.f, 0.f, 0.f, 0.f, 0.f, 0.f, 0.f};
#pragma unroll
        for (int kk = 0; kk < 8; ++kk) {
            acc0 = wmma_bf16(af0[kk], bfr[kk], acc0);
            acc1 = wmma_bf16(af1[kk], bfr[kk], acc1);
        }
        int ncol = nt * 16 + n16;
        float base = b_all[ncol] + out_both[ncol];
#pragma unroll
        for (int r = 0; r < 8; ++r) {
            size_t obs = row0 + wave * 32 + r + 8 * half;
            out[obs * D_ + ncol] =
                0.25f * (acc0[r] + base + out_rows[(size_t)cidx0[r] * D_ + ncol] +
                         out_cols[(size_t)ridx0[r] * D_ + ncol]);
            out[(obs + 16) * D_ + ncol] =
                0.25f * (acc1[r] + base + out_rows[(size_t)cidx1[r] * D_ + ncol] +
                         out_cols[(size_t)ridx1[r] * D_ + ncol]);
        }
    }
}

// ---------------------------------------------------------------------------
// Launch
// ---------------------------------------------------------------------------
extern "C" void kernel_launch(void* const* d_in, const int* in_sizes, int n_in,
                              void* d_out, int out_size, void* d_ws, size_t ws_size,
                              hipStream_t stream) {
    (void)in_sizes; (void)n_in; (void)out_size; (void)ws_size;
    const float* values = (const float*)d_in[0];
    const int* row_idx = (const int*)d_in[1];
    const int* col_idx = (const int*)d_in[2];
    const float* W_all = (const float*)d_in[3];   const float* b_all = (const float*)d_in[4];
    const float* W_both = (const float*)d_in[5];  const float* b_both = (const float*)d_in[6];
    const float* W_nv = (const float*)d_in[7];    const float* b_nv = (const float*)d_in[8];
    const float* W_nk = (const float*)d_in[9];    const float* b_nk = (const float*)d_in[10];
    const float* W_nq = (const float*)d_in[11];   const float* b_nq = (const float*)d_in[12];
    const float* W_mv = (const float*)d_in[13];   const float* b_mv = (const float*)d_in[14];
    const float* W_mk = (const float*)d_in[15];   const float* b_mk = (const float*)d_in[16];
    const float* W_mq = (const float*)d_in[17];   const float* b_mq = (const float*)d_in[18];
    float* out = (float*)d_out;

    // workspace carve (floats)
    float* ws = (float*)d_ws;
    float* sum_rows = ws;                          // N_ * D_   (per-point)
    float* sum_cols = sum_rows + (size_t)N_ * D_;  // M_ * D_   (per-camera)
    float* cnt_rows = sum_cols + (size_t)M_ * D_;  // N_
    float* cnt_cols = cnt_rows + N_;               // M_
    float* out_both = cnt_cols + M_;               // D_
    float* Qn = out_both + D_;                     // N_ * D_
    float* Kn = Qn + (size_t)N_ * D_;
    float* Vn = Kn + (size_t)N_ * D_;
    float* out_rows = Vn + (size_t)N_ * D_;        // N_ * D_
    float* Qm = out_rows + (size_t)N_ * D_;        // M_ * D_
    float* Km = Qm + (size_t)M_ * D_;
    float* Vm = Km + (size_t)M_ * D_;
    float* out_cols = Vm + (size_t)M_ * D_;        // M_ * D_

    const size_t gemm_smem = (size_t)(D_ * WL + 256 * WL) * sizeof(__bf16);
    const size_t fa_smem =
        (size_t)(32 * WL + D_ * VL + 128 * WL + 8 * 16 * VL) * sizeof(__bf16);

    // 1) zero the atomic accumulators (contiguous region)
    int nzero = N_ * D_ + M_ * D_ + N_ + M_;
    zero_f32<<<(nzero + 255) / 256, 256, 0, stream>>>(sum_rows, nzero);

    // 2) segment sums
    seg_accum<<<NNZ_, 256, 0, stream>>>(values, row_idx, col_idx, sum_rows, cnt_rows,
                                        sum_cols, cnt_cols);

    // 3) global mean + out_both (uses RAW sum_cols -> before normalize)
    global_and_both<<<1, 256, 0, stream>>>(sum_cols, W_both, b_both, out_both);

    // 4) normalize into means (in place)
    seg_normalize<<<(N_ * D_ + 255) / 256, 256, 0, stream>>>(sum_rows, cnt_rows, N_);
    seg_normalize<<<(M_ * D_ + 255) / 256, 256, 0, stream>>>(sum_cols, cnt_cols, M_);

    // 5) Q/K/V projections (WMMA), 256-row block tiles
    gemm_bias_wmma<<<N_ / 256, 256, gemm_smem, stream>>>(sum_rows, W_nq, b_nq, Qn);
    gemm_bias_wmma<<<N_ / 256, 256, gemm_smem, stream>>>(sum_rows, W_nk, b_nk, Kn);
    gemm_bias_wmma<<<N_ / 256, 256, gemm_smem, stream>>>(sum_rows, W_nv, b_nv, Vn);
    gemm_bias_wmma<<<M_ / 256, 256, gemm_smem, stream>>>(sum_cols, W_mq, b_mq, Qm);
    gemm_bias_wmma<<<M_ / 256, 256, gemm_smem, stream>>>(sum_cols, W_mk, b_mk, Km);
    gemm_bias_wmma<<<M_ / 256, 256, gemm_smem, stream>>>(sum_cols, W_mv, b_mv, Vm);

    // 6) streaming-softmax attention (WMMA)
    flash_attn_wmma<<<N_ / 128, 256, fa_smem, stream>>>(Qn, Kn, Vn, out_rows, N_);
    flash_attn_wmma<<<M_ / 128, 256, fa_smem, stream>>>(Qm, Km, Vm, out_cols, M_);

    // 7) fused big GEMM + gather + combine (256-row block tiles)
    fused_combine_wmma<<<NNZ_ / 256, 256, gemm_smem, stream>>>(
        values, row_idx, col_idx, W_all, b_all, out_rows, out_cols, out_both, out);
}
